// Edgewise_39118562132384
// MI455X (gfx1250) — matrix operand order repaired and checked by reading, not compiled, not run on hardware
//
#include <hip/hip_runtime.h>
#include <hip/hip_bf16.h>
#include <math.h>

#define NEDGES 50000
#define TILE_E 16
#define BLOCK  512
#define NW     16          // waves per block (wave32)

typedef __bf16 bf16;
typedef __attribute__((ext_vector_type(16))) __bf16 v16bf;
typedef __attribute__((ext_vector_type(8)))  __bf16 v8bf;
typedef __attribute__((ext_vector_type(8)))  float  v8f;

// reduced-sphere index tables (LMAX=4, MMAX=2)
__constant__ int c_OUT_MASK[19] = {0,1,2,3,4,5,6,7,8,10,11,12,13,14,18,19,20,21,22};
__constant__ int c_PACK_POS[19] = {0,9,1,5,16,10,2,6,13,17,11,3,7,14,18,12,4,8,15};
__constant__ int c_RAD_ROW[19] = {0,1,2,3,4, 5,6,7,8, 5,6,7,8, 9,10,11, 9,10,11};

// ---- WMMA bf16 operand index maps (wave32, contiguous k = 16*(lane/16) + h, same for A and B) ----
__device__ __forceinline__ int kA_idx(int h, int lane) {
  return ((lane >> 4) << 4) + h;
}
__device__ __forceinline__ int kB_idx(int h, int lane) {
  return ((lane >> 4) << 4) + h;
}

// A fragment: per lane two contiguous 16B runs -> 2x ds_load_b128
__device__ __forceinline__ v16bf load_a(const bf16* __restrict__ rowbase, int kbase, int lane) {
  int off = kbase + ((lane >> 4) << 4);
  v8bf lo = *(const v8bf*)(rowbase + off);
  v8bf hi = *(const v8bf*)(rowbase + off + 8);
  v16bf a;
#pragma unroll
  for (int i = 0; i < 8; ++i) { a[i] = lo[i]; a[8 + i] = hi[i]; }
  return a;
}
// B fragment from f32 weight row: 16 contiguous f32 -> 4x b128 + cvt (fallback path)
__device__ __forceinline__ v16bf load_bw(const float* __restrict__ wrow, int kbase, int lane) {
  int off = kbase + ((lane >> 4) << 4);
  v16bf b;
#pragma unroll
  for (int h = 0; h < 16; ++h) b[h] = (bf16)wrow[off + h];
  return b;
}
// B fragment from pre-converted bf16 weight row: 2x 16B loads, no cvt (fast path)
__device__ __forceinline__ v16bf load_bw(const bf16* __restrict__ wrow, int kbase, int lane) {
  int off = kbase + ((lane >> 4) << 4);
  v8bf lo = *(const v8bf*)(wrow + off);
  v8bf hi = *(const v8bf*)(wrow + off + 8);
  v16bf b;
#pragma unroll
  for (int i = 0; i < 8; ++i) { b[i] = lo[i]; b[8 + i] = hi[i]; }
  return b;
}
__device__ __forceinline__ v8f wmma_bf16(v16bf a, v16bf b, v8f c) {
  return __builtin_amdgcn_wmma_f32_16x16x32_bf16(false, a, false, b, (short)0, c, false, false);
}
__device__ __forceinline__ float sigm(float v) { return 1.f / (1.f + __expf(-v)); }

// ---- dynamic LDS layout (bytes), total 274944 ----
#define OFF_HBF 512
#define OFF_XM  (OFF_HBF + 4096)          // bf16 [16][19][256] ; later msg3 [16][19][128]
#define OFF_GAT (OFF_XM + 155648)         // bf16 [16][512]
#define OFF_R   (OFF_GAT + 16384)         // hraw f32[16][128] / rad bf16[16][3072] / m2 bf16[16][19][128]
#define SMEM_BYTES (OFF_R + 98304)

// paired-N SO2 block for m>0; complex combine + gate/env in registers.
template <typename WT>
__device__ __forceinline__ void so2_m_block(
    const bf16* __restrict__ aR, const bf16* __restrict__ aI,
    const WT* __restrict__ W, int ldb, int ksteps,
    int half, int njobs, int pR, int pI, int gl_add,
    const bf16* __restrict__ gat, const float* __restrict__ sEnv,
    bf16* __restrict__ dst, int w, int lane, int col, int hi8)
{
  for (int j = w; j < njobs; j += NW) {
    int nlo = j * 16 + col, nhi = nlo + half;
    const WT* Wlo = W + (size_t)nlo * ldb;
    const WT* Whi = W + (size_t)nhi * ldb;
    v8f crl = {}, cil = {}, crh = {}, cih = {};
    for (int ks = 0; ks < ksteps; ++ks) {
      int kb = ks * 32;
      v16bf ar = load_a(aR, kb, lane);
      v16bf ai = load_a(aI, kb, lane);
      v16bf bl = load_bw(Wlo, kb, lane);
      v16bf bh = load_bw(Whi, kb, lane);
      crl = wmma_bf16(ar, bl, crl);
      cil = wmma_bf16(ai, bl, cil);
      crh = wmma_bf16(ar, bh, crh);
      cih = wmma_bf16(ai, bh, cih);
    }
    int q = j * 16 + col, jr = q >> 7, ch = q & 127;
#pragma unroll
    for (int v = 0; v < 8; ++v) {
      int e = v + hi8;
      float vr = crl[v] - cih[v];   // yr[0] - yi[1]
      float vi = cil[v] + crh[v];   // yr[1] + yi[0]
      if (gat) {
        float g = (float)gat[e * 512 + (jr + gl_add) * 128 + ch];
        vr *= g; vi *= g;
      } else {
        float ev = sEnv[e];
        vr *= ev; vi *= ev;
      }
      dst[(e * 19 + pR + jr) * 128 + ch] = (bf16)vr;
      dst[(e * 19 + pI + jr) * 128 + ch] = (bf16)vi;
    }
  }
}

template <typename WT>
__global__ __launch_bounds__(BLOCK)
void escn_edge_kernel(
    const float* __restrict__ x,       const float* __restrict__ x_edge,
    const float* __restrict__ edist,   const float* __restrict__ wig,
    const float* __restrict__ wig_inv,
    const WT* __restrict__ W_r1, const float* __restrict__ b_r1,
    const float* __restrict__ ln_g, const float* __restrict__ ln_b,
    const WT* __restrict__ W_r2, const float* __restrict__ b_r2,
    const WT* __restrict__ W0_1, const float* __restrict__ b0_1,
    const WT* __restrict__ Wm1_1, const WT* __restrict__ Wm2_1,
    const WT* __restrict__ W0_2, const float* __restrict__ b0_2,
    const WT* __restrict__ Wm1_2, const WT* __restrict__ Wm2_2,
    const int* __restrict__ eidx, const int* __restrict__ noff_p,
    float* __restrict__ out)
{
  extern __shared__ char smem[];
  int*   sSrc = (int*)(smem + 0);
  int*   sDst = (int*)(smem + 64);
  float* sEnv = (float*)(smem + 128);
  float* sMu  = (float*)(smem + 192);
  float* sRs  = (float*)(smem + 256);
  bf16*  hbf  = (bf16*) (smem + OFF_HBF);
  bf16*  xm   = (bf16*) (smem + OFF_XM);
  bf16*  gat  = (bf16*) (smem + OFF_GAT);
  float* hraw = (float*)(smem + OFF_R);
  bf16*  radb = (bf16*) (smem + OFF_R);
  bf16*  m2   = (bf16*) (smem + OFF_R);
  bf16*  msg3 = xm;

  const int tid  = threadIdx.x;
  const int lane = tid & 31;
  const int w    = tid >> 5;
  const int col  = lane & 15;
  const int row  = lane & 15;
  const int hi8  = (lane >> 4) << 3;
  const int eg0  = blockIdx.x * TILE_E;

  __builtin_prefetch(W0_1, 0, 3);
  __builtin_prefetch(Wm1_1, 0, 3);

  // -------- P0: edge meta + x_edge staging --------
  if (tid < TILE_E) {
    int eg = eg0 + tid;
    sSrc[tid] = eidx[eg];
    sDst[tid] = eidx[NEDGES + eg];
    float d  = edist[eg] * (1.0f / 6.0f);
    float d2 = d * d;
    float d5 = d2 * d2 * d;
    float env = 1.0f + d5 * (-21.0f + d * (35.0f - 15.0f * d));
    sEnv[tid] = (d < 1.0f) ? env : 0.0f;
  }
  for (int idx = tid; idx < TILE_E * 128; idx += BLOCK) {
    int e = idx >> 7, n = idx & 127;
    hbf[idx] = (bf16)x_edge[(size_t)(eg0 + e) * 128 + n];
  }
  __syncthreads();
  // radial layer 1: h = xe @ W_r1^T + b_r1
  {
    v16bf afr[4];
#pragma unroll
    for (int ks = 0; ks < 4; ++ks) afr[ks] = load_a(hbf + row * 128, ks * 32, lane);
    for (int nt = w; nt < 8; nt += NW) {
      int n = nt * 16 + col;
      const WT* Wr = W_r1 + (size_t)n * 128;
      v8f c = {};
#pragma unroll
      for (int ks = 0; ks < 4; ++ks) c = wmma_bf16(afr[ks], load_bw(Wr, ks * 32, lane), c);
      float bias = b_r1[n];
#pragma unroll
      for (int v = 0; v < 8; ++v) hraw[(v + hi8) * 128 + n] = c[v] + bias;
    }
  }
  __syncthreads();
  if (tid < TILE_E) {  // LayerNorm stats
    float mu = 0.f;
    for (int k = 0; k < 128; ++k) mu += hraw[tid * 128 + k];
    mu *= (1.0f / 128.0f);
    float var = 0.f;
    for (int k = 0; k < 128; ++k) { float dl = hraw[tid * 128 + k] - mu; var += dl * dl; }
    var *= (1.0f / 128.0f);
    sMu[tid] = mu; sRs[tid] = rsqrtf(var + 1e-5f);
  }
  __syncthreads();
  for (int idx = tid; idx < TILE_E * 128; idx += BLOCK) {  // LN + SiLU -> hbf
    int e = idx >> 7, n = idx & 127;
    float hv = (hraw[idx] - sMu[e]) * sRs[e] * ln_g[n] + ln_b[n];
    hbf[idx] = (bf16)(hv * sigm(hv));
  }
  __syncthreads();

  // -------- P1: rad = h @ W_r2^T + b_r2 -> bf16 [16][3072] --------
  {
    v16bf afr[4];
#pragma unroll
    for (int ks = 0; ks < 4; ++ks) afr[ks] = load_a(hbf + row * 128, ks * 32, lane);
    for (int nt = w; nt < 192; nt += NW) {
      int n = nt * 16 + col;
      const WT* Wr = W_r2 + (size_t)n * 128;
      v8f c = {};
#pragma unroll
      for (int ks = 0; ks < 4; ++ks) c = wmma_bf16(afr[ks], load_bw(Wr, ks * 32, lane), c);
      float bias = b_r2[n];
#pragma unroll
      for (int v = 0; v < 8; ++v) radb[(v + hi8) * 3072 + n] = (bf16)(c[v] + bias);
    }
  }
  __syncthreads();

  // -------- P2: per-edge Wigner rotate + rad scaling -> xm (packed bf16) --------
  for (int jj = w; jj < 32; jj += NW) {
    int e = jj >> 1, mt = jj & 1;
    const float* wg = wig + (size_t)(eg0 + e) * 625;
    int srcN = sSrc[e], dstN = sDst[e];
    int r = mt * 16 + row;
    v16bf a;
#pragma unroll
    for (int h = 0; h < 16; ++h) {
      int k = kA_idx(h, lane);
      a[h] = (r < 19 && k < 25) ? (bf16)wg[c_OUT_MASK[r] * 25 + k] : (bf16)0.f;
    }
    for (int nt = 0; nt < 16; ++nt) {
      int n = nt * 16 + col;
      v16bf b;
#pragma unroll
      for (int h = 0; h < 16; ++h) {
        int k = kB_idx(h, lane);
        float bv = 0.f;
        if (k < 25)
          bv = (n < 128) ? x[(size_t)srcN * 3200 + k * 128 + n]
                         : x[(size_t)dstN * 3200 + k * 128 + (n - 128)];
        b[h] = (bf16)bv;
      }
      v8f c = {};
      c = wmma_bf16(a, b, c);
#pragma unroll
      for (int v = 0; v < 8; ++v) {
        int rr = mt * 16 + v + hi8;
        if (rr < 19) {
          int p = c_PACK_POS[rr];
          float val = c[v] * (float)radb[e * 3072 + c_RAD_ROW[p] * 256 + n];
          xm[(e * 19 + p) * 256 + n] = (bf16)val;
        }
      }
    }
  }
  __syncthreads();

  // -------- P3: SO2 conv #1 --------
  // m=0: n<512 -> gates (sigmoid now); n>=512 -> raw m0 rows into m2 (fixup after barrier)
  {
    const bf16* abase = xm + (size_t)row * (19 * 256);
    for (int j = w; j < 36; j += NW) {
      int n0 = j * 16 + col, n1 = n0 + 576;
      const WT* Wp0 = W0_1 + (size_t)n0 * 1280;
      const WT* Wp1 = W0_1 + (size_t)n1 * 1280;
      v8f c0 = {}, c1 = {};
      for (int ks = 0; ks < 40; ++ks) {
        int kb = ks * 32;
        v16bf a = load_a(abase, kb, lane);
        c0 = wmma_bf16(a, load_bw(Wp0, kb, lane), c0);
        c1 = wmma_bf16(a, load_bw(Wp1, kb, lane), c1);
      }
      float bias0 = b0_1[n0], bias1 = b0_1[n1];
#pragma unroll
      for (int v = 0; v < 8; ++v) {
        int e = v + hi8;
        float v0 = c0[v] + bias0, v1 = c1[v] + bias1;
        if (n0 < 512) gat[e * 512 + n0] = (bf16)sigm(v0);
        else { int q = n0 - 512; m2[(e * 19 + (q >> 7)) * 128 + (q & 127)] = (bf16)v0; }
        { int q = n1 - 512; m2[(e * 19 + (q >> 7)) * 128 + (q & 127)] = (bf16)v1; }
      }
    }
  }
  __syncthreads();
  for (int idx = tid; idx < TILE_E * 640; idx += BLOCK) {  // fixup m0 rows (silu / gate)
    int e = idx / 640, q = idx - e * 640;
    int p = q >> 7, ch = q & 127;
    float val = (float)m2[(e * 19 + p) * 128 + ch];
    if (p == 0) val = val * sigm(val);
    else        val = val * (float)gat[e * 512 + (p - 1) * 128 + ch];
    m2[(e * 19 + p) * 128 + ch] = (bf16)val;
  }
  so2_m_block(xm + (size_t)row * 4864 + 5 * 256, xm + (size_t)row * 4864 + 9 * 256,
              Wm1_1, 1024, 32, 512, 32, 5, 9, 0, gat, sEnv, m2, w, lane, col, hi8);
  so2_m_block(xm + (size_t)row * 4864 + 13 * 256, xm + (size_t)row * 4864 + 16 * 256,
              Wm2_1, 768, 24, 384, 24, 13, 16, 1, gat, sEnv, m2, w, lane, col, hi8);
  __syncthreads();

  // -------- P4: SO2 conv #2 (+ envelope) -> msg3 --------
  {
    const bf16* abase = m2 + (size_t)row * 2432;
    for (int j = w; j < 20; j += NW) {
      int n0 = j * 16 + col, n1 = n0 + 320;
      const WT* Wp0 = W0_2 + (size_t)n0 * 640;
      const WT* Wp1 = W0_2 + (size_t)n1 * 640;
      v8f c0 = {}, c1 = {};
      for (int ks = 0; ks < 20; ++ks) {
        int kb = ks * 32;
        v16bf a = load_a(abase, kb, lane);
        c0 = wmma_bf16(a, load_bw(Wp0, kb, lane), c0);
        c1 = wmma_bf16(a, load_bw(Wp1, kb, lane), c1);
      }
      float bias0 = b0_2[n0], bias1 = b0_2[n1];
#pragma unroll
      for (int v = 0; v < 8; ++v) {
        int e = v + hi8; float ev = sEnv[e];
        msg3[(e * 19 + (n0 >> 7)) * 128 + (n0 & 127)] = (bf16)((c0[v] + bias0) * ev);
        msg3[(e * 19 + (n1 >> 7)) * 128 + (n1 & 127)] = (bf16)((c1[v] + bias1) * ev);
      }
    }
  }
  so2_m_block(m2 + (size_t)row * 2432 + 5 * 128, m2 + (size_t)row * 2432 + 9 * 128,
              Wm1_2, 512, 16, 512, 32, 5, 9, 0, (const bf16*)nullptr, sEnv, msg3, w, lane, col, hi8);
  so2_m_block(m2 + (size_t)row * 2432 + 13 * 128, m2 + (size_t)row * 2432 + 16 * 128,
              Wm2_2, 384, 12, 384, 24, 13, 16, 0, (const bf16*)nullptr, sEnv, msg3, w, lane, col, hi8);
  __syncthreads();

  // -------- P5: inverse Wigner rotate + scatter-add --------
  {
    const int noff = noff_p[0];
    for (int jj = w; jj < 32; jj += NW) {
      int e = jj >> 1, mt = jj & 1;
      const float* wi = wig_inv + (size_t)(eg0 + e) * 625;
      int u = mt * 16 + row;
      v16bf a;
#pragma unroll
      for (int h = 0; h < 16; ++h) {
        int k = kA_idx(h, lane);
        a[h] = (u < 25 && k < 19) ? (bf16)wi[u * 25 + c_OUT_MASK[k]] : (bf16)0.f;
      }
      float* outp = out + (size_t)(sDst[e] - noff) * 3200;
      for (int nt = 0; nt < 8; ++nt) {
        int n = nt * 16 + col;
        v16bf b;
#pragma unroll
        for (int h = 0; h < 16; ++h) {
          int k = kB_idx(h, lane);
          b[h] = (k < 19) ? msg3[(e * 19 + c_PACK_POS[k]) * 128 + n] : (bf16)0.f;
        }
        v8f c = {};
        c = wmma_bf16(a, b, c);
#pragma unroll
        for (int v = 0; v < 8; ++v) {
          int uu = mt * 16 + v + hi8;
          if (uu < 25) atomicAdd(outp + uu * 128 + n, c[v]);
        }
      }
    }
  }
}

// ---- one-shot f32 -> bf16 weight conversion into d_ws (idempotent) ----
#define WO_R1  0u
#define WO_R2  16384u
#define WO_01  409600u
#define WO_M11 1884160u
#define WO_M21 2932736u
#define WO_02  3522560u
#define WO_M12 3932160u
#define WO_M22 4456448u
#define WTOT   4751360u

__global__ void convert_weights(
    const float* __restrict__ a0, const float* __restrict__ a1,
    const float* __restrict__ a2, const float* __restrict__ a3,
    const float* __restrict__ a4, const float* __restrict__ a5,
    const float* __restrict__ a6, const float* __restrict__ a7,
    bf16* __restrict__ dst)
{
  for (unsigned i = blockIdx.x * blockDim.x + threadIdx.x; i < WTOT;
       i += gridDim.x * blockDim.x) {
    float v;
    if      (i < WO_R2)  v = a0[i - WO_R1];
    else if (i < WO_01)  v = a1[i - WO_R2];
    else if (i < WO_M11) v = a2[i - WO_01];
    else if (i < WO_M21) v = a3[i - WO_M11];
    else if (i < WO_02)  v = a4[i - WO_M21];
    else if (i < WO_M12) v = a5[i - WO_02];
    else if (i < WO_M22) v = a6[i - WO_M12];
    else                 v = a7[i - WO_M22];
    dst[i] = (bf16)v;
  }
}

extern "C" void kernel_launch(void* const* d_in, const int* in_sizes, int n_in,
                              void* d_out, int out_size, void* d_ws, size_t ws_size,
                              hipStream_t stream) {
  (void)in_sizes; (void)n_in;
  hipMemsetAsync(d_out, 0, (size_t)out_size * sizeof(float), stream);

  dim3 grid(NEDGES / TILE_E);   // 3125 blocks of 16 edges
  dim3 block(BLOCK);

  if (ws_size >= (size_t)WTOT * sizeof(bf16)) {
    bf16* wsb = (bf16*)d_ws;
    convert_weights<<<1024, 256, 0, stream>>>(
        (const float*)d_in[5],  (const float*)d_in[9],  (const float*)d_in[11],
        (const float*)d_in[13], (const float*)d_in[14], (const float*)d_in[15],
        (const float*)d_in[17], (const float*)d_in[18], wsb);
    escn_edge_kernel<bf16><<<grid, block, SMEM_BYTES, stream>>>(
        (const float*)d_in[0],  (const float*)d_in[1],  (const float*)d_in[2],
        (const float*)d_in[3],  (const float*)d_in[4],
        wsb + WO_R1,  (const float*)d_in[6],  (const float*)d_in[7], (const float*)d_in[8],
        wsb + WO_R2,  (const float*)d_in[10],
        wsb + WO_01,  (const float*)d_in[12], wsb + WO_M11, wsb + WO_M21,
        wsb + WO_02,  (const float*)d_in[16], wsb + WO_M12, wsb + WO_M22,
        (const int*)d_in[19], (const int*)d_in[20],
        (float*)d_out);
  } else {
    escn_edge_kernel<float><<<grid, block, SMEM_BYTES, stream>>>(
        (const float*)d_in[0],  (const float*)d_in[1],  (const float*)d_in[2],
        (const float*)d_in[3],  (const float*)d_in[4],
        (const float*)d_in[5],  (const float*)d_in[6],  (const float*)d_in[7],  (const float*)d_in[8],
        (const float*)d_in[9],  (const float*)d_in[10],
        (const float*)d_in[11], (const float*)d_in[12], (const float*)d_in[13], (const float*)d_in[14],
        (const float*)d_in[15], (const float*)d_in[16], (const float*)d_in[17], (const float*)d_in[18],
        (const int*)d_in[19],   (const int*)d_in[20],
        (float*)d_out);
  }
}
